// RETINANET_88991722373262
// MI455X (gfx1250) — matrix operand order repaired and checked by reading, not compiled, not run on hardware
//
#include <hip/hip_runtime.h>
#include <hip/hip_bf16.h>
#include <math.h>
#include <stdint.h>

typedef __attribute__((ext_vector_type(16))) __bf16   v16bf;
typedef __attribute__((ext_vector_type(8)))  float    v8f;
typedef __attribute__((ext_vector_type(4)))  unsigned u32x4;
typedef int v4i_vs __attribute__((vector_size(16)));   // matches async-LDS builtin param

#define TOT_ANCH 76725

#if __has_builtin(__builtin_amdgcn_global_load_async_to_lds_b128) && \
    __has_builtin(__builtin_amdgcn_s_wait_asynccnt)
#define USE_ASYNC_LDS 1
#else
#define USE_ASYNC_LDS 0
#endif

__device__ __forceinline__ unsigned short f2bf(float f) {
  unsigned u = __float_as_uint(f);
  unsigned r = u + 0x7FFFu + ((u >> 16) & 1u);
  return (unsigned short)(r >> 16);
}
__device__ __forceinline__ float bf2f(unsigned short h) {
  return __uint_as_float(((unsigned)h) << 16);
}

// 16-byte global -> LDS copy: async DMA on CDNA5, register bounce otherwise.
__device__ __forceinline__ void copy16_g2l(const unsigned short* g, unsigned short* l) {
#if USE_ASYNC_LDS
  __builtin_amdgcn_global_load_async_to_lds_b128(
      (__attribute__((address_space(1))) v4i_vs*)(uintptr_t)g,
      (__attribute__((address_space(3))) v4i_vs*)l, 0, 0);
#else
  *(u32x4*)l = *(const u32x4*)g;
#endif
}
template <int N>
__device__ __forceinline__ void wait_async_le() {
#if USE_ASYNC_LDS
  __builtin_amdgcn_s_wait_asynccnt(N);   // immediate operand required
#endif
}

// ---------------- layout prep: NCHW f32 -> NHWC bf16 ----------------
__global__ void k_to_nhwc(const float* __restrict__ in, unsigned short* __restrict__ out,
                          int Bn, int C, int H, int W) {
  long long total = (long long)Bn * C * H * W;
  for (long long i = (long long)blockIdx.x * blockDim.x + threadIdx.x; i < total;
       i += (long long)gridDim.x * blockDim.x) {
    int c = (int)(i % C); long long t = i / C;
    int x = (int)(t % W); t /= W;
    int y = (int)(t % H); int b = (int)(t / H);
    out[i] = f2bf(in[(((long long)b * C + c) * H + y) * W + x]);
  }
}

// -------- weight prep: OIHW f32 -> WtT[K][Op] bf16, K = (ky*ks+kx)*C + c --------
__global__ void k_prep_wt(const float* __restrict__ w, unsigned short* __restrict__ out,
                          int O, int C, int ks, int Op) {
  long long total = (long long)ks * ks * C * Op;
  for (long long i = (long long)blockIdx.x * blockDim.x + threadIdx.x; i < total;
       i += (long long)gridDim.x * blockDim.x) {
    int o = (int)(i % Op); long long t = i / Op;
    int c = (int)(t % C);  int p = (int)(t / C);
    int ky = p / ks, kx = p % ks;
    float v = 0.f;
    if (o < O) v = w[(((long long)o * C + c) * ks + ky) * ks + kx];
    out[i] = f2bf(v);
  }
}

// -------- in-place FPN top-down: dst += nearest-2x-upsample(src) --------
__global__ void k_up2_add(unsigned short* __restrict__ dst, const unsigned short* __restrict__ src,
                          int Bn, int H, int W, int C) {
  long long total = (long long)Bn * H * W * C;
  int Hs = H >> 1, Ws = W >> 1;
  for (long long i = (long long)blockIdx.x * blockDim.x + threadIdx.x; i < total;
       i += (long long)gridDim.x * blockDim.x) {
    int c = (int)(i % C); long long t = i / C;
    int x = (int)(t % W); t /= W;
    int y = (int)(t % H); int b = (int)(t / H);
    long long si = (((long long)b * Hs + (y >> 1)) * Ws + (x >> 1)) * C + c;
    dst[i] = f2bf(bf2f(dst[i]) + bf2f(src[si]));
  }
}

// ---------------- anchors (RetinaNet P3-P7, 3 ratios x 3 scales) ----------------
__global__ void k_anchors(float* __restrict__ out) {
  int t = blockIdx.x * blockDim.x + threadIdx.x;
  if (t >= TOT_ANCH) return;
  const int offs[6] = {0, 57600, 72000, 75600, 76500, 76725};
  const int fws[5]  = {80, 40, 20, 10, 5};
  int lvl = 0;
  while (t >= offs[lvl + 1]) lvl++;
  int r = t - offs[lvl];
  int pos = r / 9, a = r % 9;
  int fw = fws[lvl];
  int y = pos / fw, x = pos % fw;
  float stride = (float)(8 << lvl);
  float cx = (x + 0.5f) * stride, cy = (y + 0.5f) * stride;
  float base = 4.f * stride;
  float scale = exp2f((float)(a % 3) * (1.0f / 3.0f));
  int ri = a / 3;
  float ratio = (ri == 0) ? 0.5f : ((ri == 1) ? 1.f : 2.f);
  float sr = sqrtf(ratio);
  float w = base * scale / sr;
  float h = base * scale * sr;
  out[t * 4 + 0] = cx - 0.5f * w;
  out[t * 4 + 1] = cy - 0.5f * h;
  out[t * 4 + 2] = cx + 0.5f * w;
  out[t * 4 + 3] = cy + 0.5f * h;
}

// ------------- implicit-GEMM conv, SAME pad, stride 1, WMMA bf16 -------------
// in   : NHWC bf16 [Bn,H,W,C]            (C % 32 == 0)
// wtT  : bf16 [K][Op], K = ks*ks*C, Op = O rounded up to 64 (zero padded)
// out  : bf16 NHWC (out_bf)  or f32 strided (out_f32):
//        index = b*obstride + ooffset + (y*W+x)*O + n
// act  : 0 = none, 1 = relu, 2 = sigmoid
// Tiling: block = 64(M) x 64(N); 4 waves split M, share one 64-wide B tile that
// is double-buffered in LDS via async-to-LDS DMA (ASYNCcnt pipeline).  Each
// wave reuses its A fragment across 4 independent v_wmma ops.
__global__ __launch_bounds__(128)
void k_conv_wmma(const unsigned short* __restrict__ in,
                 const unsigned short* __restrict__ wtT,
                 const float* __restrict__ bias,
                 unsigned short* __restrict__ out_bf,
                 float* __restrict__ out_f32,
                 int Bn, int H, int W, int C, int O, int Op, int ks, int act,
                 long long obstride, long long ooffset) {
  __shared__ unsigned short ldsB[2][32 * 64];   // K-chunk x 64N, double buffered (8KB)

  const int tid  = threadIdx.x;
  const int lane = tid & 31;
  const int wave = tid >> 5;
  const int l16  = lane & 15;
  const int half = lane >> 4;

  const int HW = H * W;
  const int M  = Bn * HW;
  const int n0 = (int)blockIdx.y << 6;               // shared 64-wide N tile
  const int m0 = ((int)blockIdx.x << 6) + (wave << 4);

  // A row for this lane (row = m0 + l16, clamped; tail rows masked at store)
  int mA = m0 + l16;
  if (mA >= M) mA = M - 1;
  const int bA = mA / HW;
  const int rA = mA - bA * HW;
  const int yA = rA / W;
  const int xA = rA - yA * W;

  const int rad = ks >> 1;
  const int NCH = C >> 5;            // 32-K chunks per kernel tap
  const int NC  = ks * ks * NCH;     // total K chunks

  // cooperative B-tile staging: thread -> (row 0..15 & row+16, 8 columns)
  const int brow = tid >> 3;         // 0..15
  const int bcol = (tid & 7) << 3;   // 0,8,...,56

  auto issueB = [&](int q, int buf) {
    const int p  = q / NCH;
    const int cb = (q - p * NCH) << 5;
    const long long kp = (long long)p * C + cb;
    const unsigned short* g0 = wtT + (kp + brow) * Op + n0 + bcol;
    copy16_g2l(g0,                 &ldsB[buf][brow * 64 + bcol]);
    copy16_g2l(g0 + 16LL * Op,     &ldsB[buf][(brow + 16) * 64 + bcol]);
  };

  union Frag { v16bf v; u32x4 u[2]; };
  v8f acc0 = {}, acc1 = {}, acc2 = {}, acc3 = {};

  issueB(0, 0);
  for (int q = 0; q < NC; ++q) {
    const int cur = q & 1;
    if (q + 1 < NC) {                 // uniform: prefetch next chunk (async DMA)
      issueB(q + 1, cur ^ 1);
      wait_async_le<2>();             // retire current chunk's 2 loads (in-order)
    } else {
      wait_async_le<0>();
    }
    __syncthreads();

    // A frag (16-bit A 16x32 layout): lanes 0-15: K {0..7,16..23}; lanes 16-31: +8
    const int p  = q / NCH;
    const int cb = (q - p * NCH) << 5;
    const int dy = p / ks - rad;
    const int dx = p - (p / ks) * ks - rad;
    const int yy = yA + dy;
    const int xx = xA + dx;
    const bool sval = (yy >= 0) && (yy < H) && (xx >= 0) && (xx < W);
    Frag a;
    if (sval) {
      const unsigned short* arow =
          in + ((long long)(bA * H + yy) * W + xx) * C + cb + half * 8;
      a.u[0] = *(const u32x4*)(arow);
      a.u[1] = *(const u32x4*)(arow + 16);
    } else {
      u32x4 z = {0u, 0u, 0u, 0u};
      a.u[0] = z; a.u[1] = z;
    }

    // B frags from LDS: lane holds one N column (l16), vector elems = K in A's order:
    // lanes 0-15: K {0..7,16..23}; lanes 16-31: +8
    const unsigned short* lb = &ldsB[cur][l16];
    Frag b0, b1, b2, b3;
    auto ldB = [&](int nb, Frag& f) {
      for (int i = 0; i < 2; ++i)
        for (int w = 0; w < 4; ++w) {
          const int k = i * 16 + half * 8 + w * 2;
          f.u[i][w] = (unsigned)lb[k * 64 + nb] | ((unsigned)lb[(k + 1) * 64 + nb] << 16);
        }
    };
    ldB(0, b0); ldB(16, b1); ldB(32, b2); ldB(48, b3);
    acc0 = __builtin_amdgcn_wmma_f32_16x16x32_bf16(false, a.v, false, b0.v, (short)0, acc0, false, false);
    acc1 = __builtin_amdgcn_wmma_f32_16x16x32_bf16(false, a.v, false, b1.v, (short)0, acc1, false, false);
    acc2 = __builtin_amdgcn_wmma_f32_16x16x32_bf16(false, a.v, false, b2.v, (short)0, acc2, false, false);
    acc3 = __builtin_amdgcn_wmma_f32_16x16x32_bf16(false, a.v, false, b3.v, (short)0, acc3, false, false);

    __syncthreads();   // all reads of ldsB[cur] done before it is refilled
  }

  // D layout: VGPR r -> M = r (lanes 0-15) / 8+r (lanes 16-31), N = lane&15
  v8f accs[4] = {acc0, acc1, acc2, acc3};
#pragma unroll
  for (int j = 0; j < 4; ++j) {
    const int n = n0 + (j << 4) + l16;
    if (n >= O) continue;
    const float bv = bias[n];
#pragma unroll
    for (int rr = 0; rr < 8; ++rr) {
      const int m = m0 + rr + half * 8;
      if (m >= M) continue;
      const int b = m / HW;
      const int rem = m - b * HW;
      float v = accs[j][rr] + bv;
      if (act == 1)      v = v > 0.f ? v : 0.f;
      else if (act == 2) v = 1.f / (1.f + __expf(-v));
      const long long oi = (long long)b * obstride + ooffset + (long long)rem * O + n;
      if (out_f32) out_f32[oi] = v;
      else         out_bf[oi]  = f2bf(v);
    }
  }
}

extern "C" void kernel_launch(void* const* d_in, const int* in_sizes, int n_in,
                              void* d_out, int out_size, void* d_ws, size_t ws_size,
                              hipStream_t stream) {
  (void)in_sizes; (void)n_in; (void)out_size; (void)ws_size;

  const float* c3    = (const float*)d_in[0];
  const float* c4    = (const float*)d_in[1];
  const float* c5    = (const float*)d_in[2];
  const float* p6    = (const float*)d_in[3];
  const float* p7    = (const float*)d_in[4];
  const float* fl1_w = (const float*)d_in[5];  const float* fl1_b = (const float*)d_in[6];
  const float* fl2_w = (const float*)d_in[7];  const float* fl2_b = (const float*)d_in[8];
  const float* fl3_w = (const float*)d_in[9];  const float* fl3_b = (const float*)d_in[10];
  const float* ftd_w = (const float*)d_in[11]; const float* ftd_b = (const float*)d_in[12];
  const float* cwv[4] = {(const float*)d_in[13], (const float*)d_in[15],
                         (const float*)d_in[17], (const float*)d_in[19]};
  const float* cbv[4] = {(const float*)d_in[14], (const float*)d_in[16],
                         (const float*)d_in[18], (const float*)d_in[20]};
  const float* chw = (const float*)d_in[21]; const float* chb = (const float*)d_in[22];
  const float* rwv[4] = {(const float*)d_in[23], (const float*)d_in[25],
                         (const float*)d_in[27], (const float*)d_in[29]};
  const float* rbv[4] = {(const float*)d_in[24], (const float*)d_in[26],
                         (const float*)d_in[28], (const float*)d_in[30]};
  const float* rhw = (const float*)d_in[31]; const float* rhb = (const float*)d_in[32];

  // ---- workspace carve-out (bf16 buffers, 256B aligned) ----
  size_t off = 0;
  auto alloc = [&](long long elems) -> unsigned short* {
    unsigned short* p = (unsigned short*)((char*)d_ws + off);
    off += (size_t)((elems * 2 + 255) & ~255LL);
    return p;
  };
  unsigned short* c3h  = alloc(2LL * 80 * 80 * 512);
  unsigned short* c4h  = alloc(2LL * 40 * 40 * 1024);
  unsigned short* c5h  = alloc(2LL * 20 * 20 * 2048);
  unsigned short* p6h  = alloc(2LL * 10 * 10 * 256);
  unsigned short* p7h  = alloc(2LL * 5 * 5 * 256);
  unsigned short* fl1t = alloc(2048LL * 256);
  unsigned short* fl2t = alloc(1024LL * 256);
  unsigned short* fl3t = alloc(512LL * 256);
  unsigned short* ftdt = alloc(9LL * 256 * 256);
  unsigned short* cwt[4]; for (int i = 0; i < 4; i++) cwt[i] = alloc(9LL * 256 * 256);
  unsigned short* rwt[4]; for (int i = 0; i < 4; i++) rwt[i] = alloc(9LL * 256 * 256);
  unsigned short* chwt = alloc(9LL * 256 * 768);   // Op = 768 (720 padded to 64)
  unsigned short* rhwt = alloc(9LL * 256 * 64);    // Op = 64  (36 padded to 64)
  unsigned short* P5l  = alloc(2LL * 20 * 20 * 256);
  unsigned short* P4l  = alloc(2LL * 40 * 40 * 256);
  unsigned short* P3l  = alloc(2LL * 80 * 80 * 256);
  unsigned short* P5f  = alloc(2LL * 20 * 20 * 256);
  unsigned short* P4f  = alloc(2LL * 40 * 40 * 256);
  unsigned short* P3f  = alloc(2LL * 80 * 80 * 256);
  unsigned short* ping = alloc(2LL * 80 * 80 * 256);
  unsigned short* pong = alloc(2LL * 80 * 80 * 256);

  auto blocks = [](long long total) { return dim3((unsigned)((total + 255) / 256)); };

  // ---- feature layout conversion ----
  k_to_nhwc<<<blocks(2LL*512*80*80),  256, 0, stream>>>(c3, c3h, 2, 512,  80, 80);
  k_to_nhwc<<<blocks(2LL*1024*40*40), 256, 0, stream>>>(c4, c4h, 2, 1024, 40, 40);
  k_to_nhwc<<<blocks(2LL*2048*20*20), 256, 0, stream>>>(c5, c5h, 2, 2048, 20, 20);
  k_to_nhwc<<<blocks(2LL*256*10*10),  256, 0, stream>>>(p6, p6h, 2, 256,  10, 10);
  k_to_nhwc<<<blocks(2LL*256*5*5),    256, 0, stream>>>(p7, p7h, 2, 256,  5,  5);

  // ---- weight transform ----
  k_prep_wt<<<blocks(2048LL*256),   256, 0, stream>>>(fl1_w, fl1t, 256, 2048, 1, 256);
  k_prep_wt<<<blocks(1024LL*256),   256, 0, stream>>>(fl2_w, fl2t, 256, 1024, 1, 256);
  k_prep_wt<<<blocks(512LL*256),    256, 0, stream>>>(fl3_w, fl3t, 256, 512,  1, 256);
  k_prep_wt<<<blocks(9LL*256*256),  256, 0, stream>>>(ftd_w, ftdt, 256, 256,  3, 256);
  for (int i = 0; i < 4; i++) {
    k_prep_wt<<<blocks(9LL*256*256), 256, 0, stream>>>(cwv[i], cwt[i], 256, 256, 3, 256);
    k_prep_wt<<<blocks(9LL*256*256), 256, 0, stream>>>(rwv[i], rwt[i], 256, 256, 3, 256);
  }
  k_prep_wt<<<blocks(9LL*256*768), 256, 0, stream>>>(chw, chwt, 720, 256, 3, 768);
  k_prep_wt<<<blocks(9LL*256*64),  256, 0, stream>>>(rhw, rhwt, 36,  256, 3, 64);

  auto conv = [&](const unsigned short* in, const unsigned short* wt, const float* bias,
                  unsigned short* obf, float* of32,
                  int H, int W, int C, int O, int Op, int ks, int act,
                  long long obstride, long long ooffset) {
    int M = 2 * H * W;
    dim3 grid((M + 63) / 64, (O + 63) / 64);   // block = 64M x 64N, 4 waves split M
    k_conv_wmma<<<grid, 128, 0, stream>>>(in, wt, bias, obf, of32,
                                          2, H, W, C, O, Op, ks, act, obstride, ooffset);
  };

  // ---- FPN ----
  conv(c5h, fl1t, fl1_b, P5l, nullptr, 20, 20, 2048, 256, 256, 1, 0, 20LL*20*256, 0);
  conv(c4h, fl2t, fl2_b, P4l, nullptr, 40, 40, 1024, 256, 256, 1, 0, 40LL*40*256, 0);
  conv(c3h, fl3t, fl3_b, P3l, nullptr, 80, 80, 512,  256, 256, 1, 0, 80LL*80*256, 0);
  k_up2_add<<<blocks(2LL*40*40*256), 256, 0, stream>>>(P4l, P5l, 2, 40, 40, 256);
  k_up2_add<<<blocks(2LL*80*80*256), 256, 0, stream>>>(P3l, P4l, 2, 80, 80, 256);
  conv(P5l, ftdt, ftd_b, P5f, nullptr, 20, 20, 256, 256, 256, 3, 0, 20LL*20*256, 0);
  conv(P4l, ftdt, ftd_b, P4f, nullptr, 40, 40, 256, 256, 256, 3, 0, 40LL*40*256, 0);
  conv(P3l, ftdt, ftd_b, P3f, nullptr, 80, 80, 256, 256, 256, 3, 0, 80LL*80*256, 0);

  // ---- heads ----
  float* cls = (float*)d_out;
  float* loc = cls + 2LL * TOT_ANCH * 80;
  float* anc = loc + 2LL * TOT_ANCH * 4;
  const unsigned short* feats[5] = {P3f, P4f, P5f, p6h, p7h};
  const int dims[5] = {80, 40, 20, 10, 5};
  const long long lvl_off[5] = {0, 57600, 72000, 75600, 76500};

  for (int l = 0; l < 5; l++) {
    int D = dims[l];
    long long fs = (long long)D * D * 256;
    // class head: 4x (conv3x3 + relu), then header -> sigmoid f32 into d_out
    conv(feats[l], cwt[0], cbv[0], ping, nullptr, D, D, 256, 256, 256, 3, 1, fs, 0);
    conv(ping,     cwt[1], cbv[1], pong, nullptr, D, D, 256, 256, 256, 3, 1, fs, 0);
    conv(pong,     cwt[2], cbv[2], ping, nullptr, D, D, 256, 256, 256, 3, 1, fs, 0);
    conv(ping,     cwt[3], cbv[3], pong, nullptr, D, D, 256, 256, 256, 3, 1, fs, 0);
    conv(pong, chwt, chb, nullptr, cls, D, D, 256, 720, 768, 3, 2,
         (long long)TOT_ANCH * 80, lvl_off[l] * 80);
    // regression head: 4x (conv3x3 + relu), then header -> raw f32 into d_out
    conv(feats[l], rwt[0], rbv[0], ping, nullptr, D, D, 256, 256, 256, 3, 1, fs, 0);
    conv(ping,     rwt[1], rbv[1], pong, nullptr, D, D, 256, 256, 256, 3, 1, fs, 0);
    conv(pong,     rwt[2], rbv[2], ping, nullptr, D, D, 256, 256, 256, 3, 1, fs, 0);
    conv(ping,     rwt[3], rbv[3], pong, nullptr, D, D, 256, 256, 256, 3, 1, fs, 0);
    conv(pong, rhwt, rhb, nullptr, loc, D, D, 256, 36, 64, 3, 0,
         (long long)TOT_ANCH * 4, lvl_off[l] * 4);
  }

  // ---- anchors ----
  k_anchors<<<(TOT_ANCH + 255) / 256, 256, 0, stream>>>(anc);
}
